// LSTM_75376676045314
// MI455X (gfx1250) — matrix operand order repaired and checked by reading, not compiled, not run on hardware
//
#include <hip/hip_runtime.h>

// ---------------- types ----------------
typedef __bf16 bf16;
typedef __attribute__((ext_vector_type(16))) __bf16 bf16x16;
typedef __attribute__((ext_vector_type(8)))  __bf16 bf16x8;
typedef __attribute__((ext_vector_type(8)))  float  f32x8;

// ---------------- problem constants ----------------
static constexpr int Vv = 32000;   // vocab
static constexpr int Ee = 512;     // embedding
static constexpr int Hh = 1024;    // hidden
static constexpr int Bb = 16;      // batch
static constexpr int Tt = 256;     // time
static constexpr int BT = Bb * Tt;     // 4096 rows
static constexpr int FH = 4 * Hh;      // 4096 (i,f,g,o)

// ---------------- fragment loaders ----------------
// A fragment (16x32 bf16, MxK): lane r=lane&15 is row M=r.
//   lanes 0-15 hold K = 0..15; lanes 16-31 hold K = 16..31.
// rowp = &A[row*K + k0]
__device__ __forceinline__ bf16x16 load_a_frag(const bf16* __restrict__ rowp, int kh) {
    const bf16x8 lo = *(const bf16x8*)(rowp + kh * 16);
    const bf16x8 hi = *(const bf16x8*)(rowp + kh * 16 + 8);
    bf16x16 r;
#pragma unroll
    for (int i = 0; i < 8; ++i) { r[i] = lo[i]; r[i + 8] = hi[i]; }
    return r;
}

// B fragment (32x16 bf16, KxN), B stored transposed as BT[N,K] row-major.
//   lane n=lane&15 is column N=n; lanes 0-15 hold K=0..15, lanes 16-31 K=16..31.
// colp = &BT[n*K + k0]
__device__ __forceinline__ bf16x16 load_b_frag(const bf16* __restrict__ colp, int kh) {
    return *(const bf16x16*)(colp + kh * 16);
}

__device__ __forceinline__ f32x8 wmma_bf16(bf16x16 a, bf16x16 b, f32x8 c) {
    return __builtin_amdgcn_wmma_f32_16x16x32_bf16(
        /*neg_a=*/false, a, /*neg_b=*/false, b,
        /*c_mod=*/(short)0, c, /*reuse_a=*/false, /*reuse_b=*/false);
}

__device__ __forceinline__ float sigmoidf_(float x) {
    return 1.0f / (1.0f + __expf(-x));
}

// ---------------- prep kernels ----------------
__global__ void zero_state_kernel(bf16* __restrict__ h, float* __restrict__ c, int n) {
    int i = blockIdx.x * blockDim.x + threadIdx.x;
    if (i < n) { h[i] = (bf16)0.0f; c[i] = 0.0f; }
}

__global__ void embed_gather_kernel(const int* __restrict__ x, const float* __restrict__ emb,
                                    bf16* __restrict__ e, int rows, int E) {
    int i = blockIdx.x * blockDim.x + threadIdx.x;
    if (i >= rows * E) return;
    int row = i / E, col = i - row * E;
    e[i] = (bf16)emb[(size_t)x[row] * E + col];
}

// in: [K,N] fp32 row-major  ->  out: [N,K] bf16 row-major
// 32x32 LDS-tiled transpose, both global sides coalesced. Requires K%32==0, N%32==0.
// grid = (K/32, N/32), block = 256 (32x8).
__global__ void __launch_bounds__(256)
transpose_to_bf16_kernel(const float* __restrict__ in, bf16* __restrict__ out,
                         int K, int N) {
    __shared__ float tile[32][33];  // +1 pad: conflict-free transposed reads
    const int kt = blockIdx.x * 32;
    const int nt = blockIdx.y * 32;
    const int tx = threadIdx.x & 31;
    const int ty = threadIdx.x >> 5;   // 0..7
#pragma unroll
    for (int j = 0; j < 4; ++j) {
        const int k = kt + ty + 8 * j;
        tile[ty + 8 * j][tx] = in[(size_t)k * N + nt + tx];   // coalesced in n
    }
    __syncthreads();
#pragma unroll
    for (int j = 0; j < 4; ++j) {
        const int n = nt + ty + 8 * j;
        out[(size_t)n * K + kt + tx] = (bf16)tile[tx][ty + 8 * j];  // coalesced in k
    }
}

// ---------------- big GEMM: C[M,N] = A[M,K](bf16) * BT[N,K]^T(bf16) + bias[N] ----------------
// block = 256 threads = 8 waves as 2(M) x 4(N).
// Each wave register-blocks 2 M-tiles x 4 N-tiles = 32x64 output (8 WMMA accumulators),
// so one k-step = 2 A-frag + 4 B-frag loads feeding 8 WMMAs (24 B/lane per WMMA).
// Block tile = 64(M) x 256(N). grid.x = N/256, grid.y = M/64.
// Requires M%64==0, N%256==0, K%32==0.
__global__ void __launch_bounds__(256)
gemm_bf16_kernel(const bf16* __restrict__ A, const bf16* __restrict__ BT,
                 const float* __restrict__ bias, float* __restrict__ C,
                 int M, int N, int K) {
    const int lane  = threadIdx.x & 31;
    const int wave  = threadIdx.x >> 5;
    const int waveM = wave >> 2;          // 0..1
    const int waveN = wave & 3;           // 0..3
    const int tileM0 = blockIdx.y * 4 + waveM * 2;        // two 16-row tiles
    const int tileN0 = (blockIdx.x * 4 + waveN) * 4;      // four 16-col tiles

    const int r  = lane & 15;
    const int kh = lane >> 4;

    const bf16* Arow0 = A + (size_t)(tileM0 * 16 + r) * K;
    const bf16* Arow1 = Arow0 + (size_t)16 * K;
    const bf16* Bp0 = BT + (size_t)((tileN0 + 0) * 16 + r) * K;
    const bf16* Bp1 = BT + (size_t)((tileN0 + 1) * 16 + r) * K;
    const bf16* Bp2 = BT + (size_t)((tileN0 + 2) * 16 + r) * K;
    const bf16* Bp3 = BT + (size_t)((tileN0 + 3) * 16 + r) * K;

    f32x8 acc00 = {}, acc01 = {}, acc02 = {}, acc03 = {};
    f32x8 acc10 = {}, acc11 = {}, acc12 = {}, acc13 = {};
#pragma unroll 2
    for (int k0 = 0; k0 < K; k0 += 32) {
        const bf16x16 a0 = load_a_frag(Arow0 + k0, kh);
        const bf16x16 a1 = load_a_frag(Arow1 + k0, kh);
        const bf16x16 b0 = load_b_frag(Bp0 + k0, kh);
        const bf16x16 b1 = load_b_frag(Bp1 + k0, kh);
        const bf16x16 b2 = load_b_frag(Bp2 + k0, kh);
        const bf16x16 b3 = load_b_frag(Bp3 + k0, kh);
        acc00 = wmma_bf16(a0, b0, acc00);
        acc10 = wmma_bf16(a1, b0, acc10);
        acc01 = wmma_bf16(a0, b1, acc01);
        acc11 = wmma_bf16(a1, b1, acc11);
        acc02 = wmma_bf16(a0, b2, acc02);
        acc12 = wmma_bf16(a1, b2, acc12);
        acc03 = wmma_bf16(a0, b3, acc03);
        acc13 = wmma_bf16(a1, b3, acc13);
    }

    // C/D layout: VGPR v holds (m = v + 8*kh, n = lane&15)
    const int n = lane & 15;
    f32x8 accs[2][4] = {{acc00, acc01, acc02, acc03},
                        {acc10, acc11, acc12, acc13}};
#pragma unroll
    for (int mi = 0; mi < 2; ++mi) {
#pragma unroll
        for (int f = 0; f < 4; ++f) {
            const int nc = (tileN0 + f) * 16 + n;
            const float bv = bias ? bias[nc] : 0.0f;
#pragma unroll
            for (int v = 0; v < 8; ++v) {
                const int mr = (tileM0 + mi) * 16 + v + 8 * kh;
                C[(size_t)mr * N + nc] = accs[mi][f][v] + bv;
            }
        }
    }
}

// ---------------- LSTM recurrent step ----------------
// One 16(B) x 16 tile per gate per wave; 64 waves cover H=1024.
// z = zx[:,t,:] + h_prev @ rec_kernel ; gates i,f,g,o ; update c,h.
__global__ void __launch_bounds__(256)
lstm_step_kernel(const bf16* __restrict__ hprev,  // [16,H] bf16
                 const bf16* __restrict__ rT,     // [4H,H] bf16 (transposed rec_kernel)
                 const float* __restrict__ zx,    // [BT,4H] fp32 (bias already added)
                 float* __restrict__ c,           // [16,H] fp32 state
                 bf16* __restrict__ hcur,         // [16,H] bf16 (next-step A)
                 bf16* __restrict__ hseq,         // [B,T,H] bf16 (output-GEMM A)
                 int t) {
    const int lane  = threadIdx.x & 31;
    const int gwave = (blockIdx.x * blockDim.x + threadIdx.x) >> 5;
    const int n0 = gwave * 16;            // hidden-column tile base
    if (n0 >= Hh) return;

    const int r  = lane & 15;
    const int kh = lane >> 4;
    const bf16* Arow = hprev + (size_t)r * Hh;

    f32x8 acc[4] = {};
    for (int k0 = 0; k0 < Hh; k0 += 32) {
        const bf16x16 a = load_a_frag(Arow + k0, kh);
#pragma unroll
        for (int g = 0; g < 4; ++g) {
            const bf16* Bb = rT + (size_t)(g * Hh + n0 + r) * Hh;
            acc[g] = wmma_bf16(a, load_b_frag(Bb + k0, kh), acc[g]);
        }
    }

    const int n = lane & 15;
    const int col = n0 + n;
#pragma unroll
    for (int v = 0; v < 8; ++v) {
        const int m = v + 8 * kh;                       // batch index
        const size_t zrow = (size_t)(m * Tt + t) * FH;
        const float zi = acc[0][v] + zx[zrow + 0 * Hh + col];
        const float zf = acc[1][v] + zx[zrow + 1 * Hh + col];
        const float zg = acc[2][v] + zx[zrow + 2 * Hh + col];
        const float zo = acc[3][v] + zx[zrow + 3 * Hh + col];
        const float ig = sigmoidf_(zi);
        const float fg = sigmoidf_(zf);
        const float gg = tanhf(zg);
        const float og = sigmoidf_(zo);
        const size_t ci = (size_t)m * Hh + col;
        const float cc = fg * c[ci] + ig * gg;
        c[ci] = cc;
        const float hh = og * tanhf(cc);
        const bf16 hb = (bf16)hh;
        hcur[ci] = hb;
        hseq[((size_t)m * Tt + t) * Hh + col] = hb;
    }
}

// ---------------- launcher ----------------
extern "C" void kernel_launch(void* const* d_in, const int* in_sizes, int n_in,
                              void* d_out, int out_size, void* d_ws, size_t ws_size,
                              hipStream_t stream) {
    (void)in_sizes; (void)n_in; (void)out_size; (void)ws_size;

    const int*   x          = (const int*)  d_in[0];   // [B,T]
    const float* emb        = (const float*)d_in[1];   // [V,E]
    const float* kernel     = (const float*)d_in[2];   // [E,4H]
    const float* rec_kernel = (const float*)d_in[3];   // [H,4H]
    const float* bias       = (const float*)d_in[4];   // [4H]
    const float* w_out      = (const float*)d_in[5];   // [H,V]
    const float* b_out      = (const float*)d_in[6];   // [V]
    float* out = (float*)d_out;                        // [B,T,V]

    // carve workspace
    char* ws = (char*)d_ws;
    size_t off = 0;
    auto carve = [&](size_t bytes) -> void* {
        void* p = ws + off;
        off += (bytes + 255) & ~(size_t)255;
        return p;
    };
    bf16*  e_bf  = (bf16*) carve((size_t)BT * Ee * sizeof(bf16));   // [BT,E]
    bf16*  kT    = (bf16*) carve((size_t)FH * Ee * sizeof(bf16));   // [4H,E]
    bf16*  rT    = (bf16*) carve((size_t)FH * Hh * sizeof(bf16));   // [4H,H]
    bf16*  wT    = (bf16*) carve((size_t)Vv * Hh * sizeof(bf16));   // [V,H]
    float* zx    = (float*)carve((size_t)BT * FH * sizeof(float));  // [BT,4H]
    bf16*  hcur  = (bf16*) carve((size_t)Bb * Hh * sizeof(bf16));   // [16,H]
    float* cst   = (float*)carve((size_t)Bb * Hh * sizeof(float));  // [16,H]
    bf16*  hseq  = (bf16*) carve((size_t)BT * Hh * sizeof(bf16));   // [B,T,H]

    const int TPB = 256;

    // 1) zero h0/c0
    zero_state_kernel<<<(Bb * Hh + TPB - 1) / TPB, TPB, 0, stream>>>(hcur, cst, Bb * Hh);

    // 2) embedding gather -> bf16
    embed_gather_kernel<<<((size_t)BT * Ee + TPB - 1) / TPB, TPB, 0, stream>>>(x, emb, e_bf, BT, Ee);

    // 3) weight transposes -> bf16 [N,K]  (LDS-tiled, coalesced both sides)
    {
        dim3 g1(Ee / 32, FH / 32);
        transpose_to_bf16_kernel<<<g1, TPB, 0, stream>>>(kernel, kT, Ee, FH);
        dim3 g2(Hh / 32, FH / 32);
        transpose_to_bf16_kernel<<<g2, TPB, 0, stream>>>(rec_kernel, rT, Hh, FH);
        dim3 g3(Hh / 32, Vv / 32);
        transpose_to_bf16_kernel<<<g3, TPB, 0, stream>>>(w_out, wT, Hh, Vv);
    }

    // 4) zx = e @ kernel + bias : [4096,4096] = [4096,512]x[512,4096]
    {
        dim3 grid(FH / 256, BT / 64);
        gemm_bf16_kernel<<<grid, TPB, 0, stream>>>(e_bf, kT, bias, zx, BT, FH, Ee);
    }

    // 5) sequential LSTM scan: 64 waves = 8 blocks x 256 threads
    for (int t = 0; t < Tt; ++t) {
        lstm_step_kernel<<<(Hh / 16) / 8, TPB, 0, stream>>>(hcur, rT, zx, cst, hcur, hseq, t);
    }

    // 6) logits = h_seq @ w_out + b_out : [4096,32000] = [4096,1024]x[1024,32000]
    {
        dim3 grid(Vv / 256, BT / 64);
        gemm_bf16_kernel<<<grid, TPB, 0, stream>>>(hseq, wT, b_out, out, BT, Vv, Hh);
    }
}
